// GroupedQueryAttention_23776938950946
// MI455X (gfx1250) — matrix operand order repaired and checked by reading, not compiled, not run on hardware
//
#include <hip/hip_runtime.h>

// ---------------- problem constants ----------------
constexpr int B_  = 2;
constexpr int L_  = 4096;
constexpr int D_  = 1024;
constexpr int NH_ = 16;
constexpr int NKV_= 4;
constexpr int HD_ = 64;
constexpr int NQK = NH_ * HD_;          // 1024
constexpr int NKK = NKV_ * HD_;         // 256
constexpr int NQKV = NQK + 2 * NKK;     // 1536
constexpr int M_  = B_ * L_;            // 8192 rows

// ---------------- types ----------------
typedef __bf16 bf16_t;
typedef __bf16 v16bf __attribute__((ext_vector_type(16)));
typedef __bf16 v8bf  __attribute__((ext_vector_type(8)));
typedef float  v8f   __attribute__((ext_vector_type(8)));
typedef int    v4i   __attribute__((vector_size(16)));

// CDNA5 async global->LDS path (ASYNCcnt), guarded so absence cannot break the build.
#if __has_builtin(__builtin_amdgcn_global_load_async_to_lds_b128) && \
    __has_builtin(__builtin_amdgcn_s_wait_asynccnt)
#define USE_ASYNC_LDS 1
#else
#define USE_ASYNC_LDS 0
#endif

// builtin signature (from hipcc diagnostic): (v4i AS1* src, v4i AS3* dst, imm, imm)
#define TO_GLB(p) ((__attribute__((address_space(1))) v4i*)(p))
#define TO_LDS(p) ((__attribute__((address_space(3))) v4i*)(p))

__device__ __forceinline__ v16bf join16(v8bf lo, v8bf hi) {
  return __builtin_shufflevector(lo, hi, 0,1,2,3,4,5,6,7,8,9,10,11,12,13,14,15);
}

__device__ __forceinline__ v8f wmma_bf16(v16bf a, v16bf b, v8f c) {
  // D = A(16x32 bf16) * B(32x16 bf16) + C(16x16 f32)
  return __builtin_amdgcn_wmma_f32_16x16x32_bf16(false, a, false, b, (short)0, c,
                                                 false, false);
}

// ---------------- conversion / packing ----------------
__global__ __launch_bounds__(256) void cvt_f32_bf16(const float* __restrict__ src,
                                                    bf16_t* __restrict__ dst, int n) {
  int i = blockIdx.x * 256 + threadIdx.x;
  if (i < n) dst[i] = (bf16_t)src[i];
}

// wqkv_t[n][k]  (n in [0,1536), k in [0,1024))  = W^T concatenated (q | k | v)
__global__ __launch_bounds__(256) void pack_wqkv_t(const float* __restrict__ wq,
                                                   const float* __restrict__ wk,
                                                   const float* __restrict__ wv,
                                                   bf16_t* __restrict__ wt) {
  int idx = blockIdx.x * 256 + threadIdx.x;
  if (idx >= NQKV * D_) return;
  int n = idx >> 10;          // / 1024
  int kk = idx & (D_ - 1);
  float v;
  if (n < NQK)            v = wq[(size_t)kk * NQK + n];
  else if (n < NQK + NKK) v = wk[(size_t)kk * NKK + (n - NQK)];
  else                    v = wv[(size_t)kk * NKK + (n - NQK - NKK)];
  wt[idx] = (bf16_t)v;
}

// wo_t[n][k] = wo[k][n]
__global__ __launch_bounds__(256) void pack_wo_t(const float* __restrict__ wo,
                                                 bf16_t* __restrict__ wt) {
  int idx = blockIdx.x * 256 + threadIdx.x;
  if (idx >= D_ * D_) return;
  int n = idx >> 10;
  int kk = idx & (D_ - 1);
  wt[idx] = (bf16_t)wo[(size_t)kk * D_ + n];
}

// ---------------- shared GEMM core ----------------
// Block = 128 threads (4 waves). All waves share the 64(N) x 32(K) B tile,
// double-buffered in LDS via async global->LDS copies; each wave owns a
// 16(M) x 64(N) accumulator tile. K = D_ = 1024 for both GEMMs.
//
// sbuf: 2 * 64 * 32 bf16 (8 KB).  arow: A row (m0 + r16) base pointer.
// bt:   B^T base + n0 row (row stride D_).
__device__ __forceinline__ void gemm_core(const bf16_t* __restrict__ arow,
                                          const bf16_t* __restrict__ bt,
                                          bf16_t* sbuf, int tid, int r16, int hf,
                                          v8f acc[4]) {
  auto stage = [&](int k, int sel) {
    bf16_t* base = sbuf + sel * (64 * 32);
#pragma unroll
    for (int t = 0; t < 2; ++t) {
      int idx  = tid + t * 128;        // 256 chunks of 16B = 64 rows x 32 K
      int row  = idx >> 2;
      int part = idx & 3;
      const bf16_t* src = bt + (size_t)row * D_ + k + part * 8;
      bf16_t*       dst = base + row * 32 + part * 8;
#if USE_ASYNC_LDS
      __builtin_amdgcn_global_load_async_to_lds_b128(TO_GLB(src), TO_LDS(dst), 0, 0);
#else
      *(uint4*)dst = *(const uint4*)src;
#endif
    }
  };

  stage(0, 0);
#if USE_ASYNC_LDS
  __builtin_amdgcn_s_wait_asynccnt(0);
#endif
  __syncthreads();

  for (int k = 0; k < D_; k += 32) {
    int sel = (k >> 5) & 1;
    if (k + 32 < D_) stage(k + 32, sel ^ 1);   // prefetch next tile (async)

    __builtin_prefetch(arow + k + 128, 0, 0);
    v16bf a = join16(*(const v8bf*)(arow + k + hf * 16),
                     *(const v8bf*)(arow + k + hf * 16 + 8));
    const bf16_t* bb = sbuf + sel * (64 * 32);
#pragma unroll
    for (int c = 0; c < 4; ++c) {
      const bf16_t* lb = bb + (c * 16 + r16) * 32 + hf * 16;
      v16bf b = join16(*(const v8bf*)(lb), *(const v8bf*)(lb + 8));
      acc[c] = wmma_bf16(a, b, acc[c]);
    }

#if USE_ASYNC_LDS
    __builtin_amdgcn_s_wait_asynccnt(0);       // next buffer resident
#endif
    __syncthreads();
  }
}

// ---------------- fused QKV GEMM + RoPE ----------------
__global__ __launch_bounds__(128) void qkv_rope_kernel(const bf16_t* __restrict__ xb,
                                                       const bf16_t* __restrict__ wt,
                                                       const float* __restrict__ freqs,
                                                       bf16_t* __restrict__ qh,
                                                       bf16_t* __restrict__ kh,
                                                       bf16_t* __restrict__ vt) {
  __shared__ bf16_t sbuf[2 * 64 * 32];
  const int tid  = threadIdx.x;
  const int lane = tid & 31;
  const int wid  = tid >> 5;
  const int r16  = lane & 15;
  const int hf   = lane >> 4;
  const int m0   = (blockIdx.x * 4 + wid) * 16;
  const int n0   = blockIdx.y * 64;

  v8f acc[4];
#pragma unroll
  for (int c = 0; c < 4; ++c)
#pragma unroll
    for (int i = 0; i < 8; ++i) acc[c][i] = 0.0f;

  gemm_core(xb + (size_t)(m0 + r16) * D_, wt + (size_t)n0 * D_,
            sbuf, tid, r16, hf, acc);

  // epilogue: C layout -> lane owns (row = r + 8*hf, col = c*16 + r16)
#pragma unroll
  for (int r = 0; r < 8; ++r) {
    int gr  = m0 + r + 8 * hf;      // in [0, B*L)
    int bb  = gr >> 12;             // / 4096
    int pos = gr & (L_ - 1);
    if (n0 < NQK) {                 // ---- Q head, apply RoPE ----
      int h = n0 / HD_;
      bf16_t* dst = qh + ((size_t)(bb * NH_ + h) * L_ + pos) * HD_;
#pragma unroll
      for (int ch = 0; ch < 2; ++ch) {
        int i = ch * 16 + r16;                  // rotary index 0..31
        float f  = freqs[pos * (HD_ / 2) + i];
        float cs = __cosf(f), sn = __sinf(f);
        float re = acc[ch][r], im = acc[ch + 2][r];
        dst[i]      = (bf16_t)(re * cs - im * sn);
        dst[i + 32] = (bf16_t)(re * sn + im * cs);
      }
    } else if (n0 < NQK + NKK) {    // ---- K head, apply RoPE ----
      int kv = (n0 - NQK) / HD_;
      bf16_t* dst = kh + ((size_t)(bb * NKV_ + kv) * L_ + pos) * HD_;
#pragma unroll
      for (int ch = 0; ch < 2; ++ch) {
        int i = ch * 16 + r16;
        float f  = freqs[pos * (HD_ / 2) + i];
        float cs = __cosf(f), sn = __sinf(f);
        float re = acc[ch][r], im = acc[ch + 2][r];
        dst[i]      = (bf16_t)(re * cs - im * sn);
        dst[i + 32] = (bf16_t)(re * sn + im * cs);
      }
    } else {                        // ---- V head: store transposed [HD][L] ----
      int kv = (n0 - NQK - NKK) / HD_;
#pragma unroll
      for (int ch = 0; ch < 4; ++ch) {
        int hd = ch * 16 + r16;
        vt[((size_t)(bb * NKV_ + kv) * HD_ + hd) * L_ + pos] = (bf16_t)acc[ch][r];
      }
    }
  }
}

// ---------------- flash attention (causal, GQA) ----------------
// one wave per (b, h, 16-query tile); 32 keys per iteration.
__global__ __launch_bounds__(32) void attn_kernel(const bf16_t* __restrict__ qh,
                                                  const bf16_t* __restrict__ kh,
                                                  const bf16_t* __restrict__ vt,
                                                  bf16_t* __restrict__ ao) {
  const int lane = threadIdx.x & 31;
  const int r16  = lane & 15;
  const int hf   = lane >> 4;
  const int q0   = blockIdx.x * 16;
  const int h    = blockIdx.y;
  const int b    = blockIdx.z;
  const int kv   = h >> 2;                 // NH/NKV = 4

  __shared__ bf16_t pb[16 * 32];           // P tile staging, C-layout -> A-layout

  // Q fragments (A layout: lane = row q0+r16; K-chunks per half)
  const bf16_t* qrow = qh + ((size_t)(b * NH_ + h) * L_ + q0 + r16) * HD_;
  v16bf qa0 = join16(*(const v8bf*)(qrow + hf * 16),
                     *(const v8bf*)(qrow + hf * 16 + 8));
  v16bf qa1 = join16(*(const v8bf*)(qrow + 32 + hf * 16),
                     *(const v8bf*)(qrow + 40 + hf * 16));

  v8f o[4];
#pragma unroll
  for (int c = 0; c < 4; ++c)
#pragma unroll
    for (int i = 0; i < 8; ++i) o[c][i] = 0.0f;

  float mrow[8], lrow[8];
#pragma unroll
  for (int r = 0; r < 8; ++r) { mrow[r] = -3.0e38f; lrow[r] = 0.0f; }

  const bf16_t* kbase = kh + (size_t)(b * NKV_ + kv) * L_ * HD_;
  const bf16_t* vbase = vt + (size_t)(b * NKV_ + kv) * HD_ * L_;
  const float SC = 0.18033688f;            // (1/sqrt(64)) * log2(e)

  for (int j = 0; j < q0 + 16; j += 32) {
    // ---- scores S = Q @ K^T for 2 x 16-key tiles ----
    v8f s[2];
#pragma unroll
    for (int kt = 0; kt < 2; ++kt) {
      const bf16_t* krow = kbase + (size_t)(j + kt * 16 + r16) * HD_ + hf * 16;
      v16bf kb0 = join16(*(const v8bf*)(krow),      *(const v8bf*)(krow + 8));
      v16bf kb1 = join16(*(const v8bf*)(krow + 32), *(const v8bf*)(krow + 40));
      v8f sc;
#pragma unroll
      for (int i = 0; i < 8; ++i) sc[i] = 0.0f;
      sc = wmma_bf16(qa0, kb0, sc);
      sc = wmma_bf16(qa1, kb1, sc);
      s[kt] = sc;
    }

    // ---- online softmax (per row; rows live in 16-lane halves) ----
    float p0v[8], p1v[8];
#pragma unroll
    for (int r = 0; r < 8; ++r) {
      int row = q0 + r + 8 * hf;
      bool v0 = (j + r16)      <= row;
      bool v1 = (j + 16 + r16) <= row;
      float s0 = v0 ? s[0][r] * SC : -3.0e38f;
      float s1 = v1 ? s[1][r] * SC : -3.0e38f;
      float mx = fmaxf(s0, s1);
#pragma unroll
      for (int off = 1; off < 16; off <<= 1) mx = fmaxf(mx, __shfl_xor(mx, off, 32));
      float mn = fmaxf(mrow[r], mx);
      float al = exp2f(mrow[r] - mn);
      float p0 = v0 ? exp2f(s0 - mn) : 0.0f;
      float p1 = v1 ? exp2f(s1 - mn) : 0.0f;
      float sum = p0 + p1;
#pragma unroll
      for (int off = 1; off < 16; off <<= 1) sum += __shfl_xor(sum, off, 32);
      lrow[r] = lrow[r] * al + sum;
      mrow[r] = mn;
#pragma unroll
      for (int c = 0; c < 4; ++c) o[c][r] *= al;
      p0v[r] = p0; p1v[r] = p1;
    }

    // ---- stage P (16x32) through LDS: C layout -> A layout ----
    __syncthreads();
#pragma unroll
    for (int r = 0; r < 8; ++r) {
      pb[(r + 8 * hf) * 32 + r16]      = (bf16_t)p0v[r];
      pb[(r + 8 * hf) * 32 + 16 + r16] = (bf16_t)p1v[r];
    }
    __syncthreads();
    v16bf pa = join16(*(const v8bf*)(&pb[r16 * 32 + hf * 16]),
                      *(const v8bf*)(&pb[r16 * 32 + hf * 16 + 8]));

    // ---- O += P @ V  (V stored transposed: contiguous B-fragments) ----
#pragma unroll
    for (int c = 0; c < 4; ++c) {
      const bf16_t* vrow = vbase + (size_t)(c * 16 + r16) * L_ + j + hf * 16;
      v16bf vb = join16(*(const v8bf*)(vrow), *(const v8bf*)(vrow + 8));
      o[c] = wmma_bf16(pa, vb, o[c]);
    }
  }

  // ---- normalize and store (bf16, row-major [B*L][NH*HD]) ----
#pragma unroll
  for (int r = 0; r < 8; ++r) {
    float inv = 1.0f / lrow[r];
    int row = q0 + r + 8 * hf;
    bf16_t* dst = ao + ((size_t)(b * L_ + row)) * NQK + h * HD_;
#pragma unroll
    for (int c = 0; c < 4; ++c) dst[c * 16 + r16] = (bf16_t)(o[c][r] * inv);
  }
}

// ---------------- output projection: out = ao @ wo (f32 out) ----------------
__global__ __launch_bounds__(128) void out_proj_kernel(const bf16_t* __restrict__ ao,
                                                       const bf16_t* __restrict__ wot,
                                                       float* __restrict__ out) {
  __shared__ bf16_t sbuf[2 * 64 * 32];
  const int tid  = threadIdx.x;
  const int lane = tid & 31;
  const int wid  = tid >> 5;
  const int r16  = lane & 15;
  const int hf   = lane >> 4;
  const int m0   = (blockIdx.x * 4 + wid) * 16;
  const int n0   = blockIdx.y * 64;

  v8f acc[4];
#pragma unroll
  for (int c = 0; c < 4; ++c)
#pragma unroll
    for (int i = 0; i < 8; ++i) acc[c][i] = 0.0f;

  gemm_core(ao + (size_t)(m0 + r16) * D_, wot + (size_t)n0 * D_,
            sbuf, tid, r16, hf, acc);

#pragma unroll
  for (int r = 0; r < 8; ++r) {
    int gr = m0 + r + 8 * hf;
    float* dst = out + (size_t)gr * D_ + n0;
#pragma unroll
    for (int c = 0; c < 4; ++c)
      __builtin_nontemporal_store(acc[c][r], dst + c * 16 + r16);  // write-once f32
  }
}

// ---------------- workspace layout (bytes) ----------------
constexpr size_t OFF_XB   = 0;                                   // x bf16       16 MB
constexpr size_t OFF_WQKV = OFF_XB   + (size_t)M_ * D_ * 2;      // wqkv^T bf16   3 MB
constexpr size_t OFF_WOT  = OFF_WQKV + (size_t)NQKV * D_ * 2;    // wo^T bf16     2 MB
constexpr size_t OFF_Q    = OFF_WOT  + (size_t)D_ * D_ * 2;      // q bf16       16 MB
constexpr size_t OFF_K    = OFF_Q    + (size_t)B_ * NH_  * L_ * HD_ * 2; // 4 MB
constexpr size_t OFF_VT   = OFF_K    + (size_t)B_ * NKV_ * L_ * HD_ * 2; // 4 MB
constexpr size_t OFF_AO   = OFF_VT   + (size_t)B_ * NKV_ * L_ * HD_ * 2; // 16 MB

extern "C" void kernel_launch(void* const* d_in, const int* in_sizes, int n_in,
                              void* d_out, int out_size, void* d_ws, size_t ws_size,
                              hipStream_t stream) {
  const float* x     = (const float*)d_in[0];
  const float* freqs = (const float*)d_in[1];
  // d_in[2] = dense causal mask: intentionally unused (causality applied analytically)
  const float* wq    = (const float*)d_in[3];
  const float* wk    = (const float*)d_in[4];
  const float* wv    = (const float*)d_in[5];
  const float* wo    = (const float*)d_in[6];

  char* ws = (char*)d_ws;
  bf16_t* xb    = (bf16_t*)(ws + OFF_XB);
  bf16_t* wqkvt = (bf16_t*)(ws + OFF_WQKV);
  bf16_t* wot   = (bf16_t*)(ws + OFF_WOT);
  bf16_t* qb    = (bf16_t*)(ws + OFF_Q);
  bf16_t* kb    = (bf16_t*)(ws + OFF_K);
  bf16_t* vtb   = (bf16_t*)(ws + OFF_VT);
  bf16_t* aob   = (bf16_t*)(ws + OFF_AO);

  // 1) precision conversion / weight packing
  cvt_f32_bf16<<<(M_ * D_ + 255) / 256, 256, 0, stream>>>(x, xb, M_ * D_);
  pack_wqkv_t<<<(NQKV * D_ + 255) / 256, 256, 0, stream>>>(wq, wk, wv, wqkvt);
  pack_wo_t<<<(D_ * D_ + 255) / 256, 256, 0, stream>>>(wo, wot);

  // 2) fused QKV projection + RoPE (WMMA, async-LDS double-buffered B tiles)
  qkv_rope_kernel<<<dim3(M_ / 64, NQKV / 64), 128, 0, stream>>>(
      xb, wqkvt, freqs, qb, kb, vtb);

  // 3) causal flash attention (WMMA QK^T and PV, online softmax)
  attn_kernel<<<dim3(L_ / 16, NH_, B_), 32, 0, stream>>>(qb, kb, vtb, aob);

  // 4) output projection (WMMA), f32 result, non-temporal stores
  out_proj_kernel<<<dim3(M_ / 64, D_ / 64), 128, 0, stream>>>(
      aob, wot, (float*)d_out);
}
